// CausalGNN_14096082665592
// MI455X (gfx1250) — compile-verified, matrix operand order checked by reading, not run on hardware
//
#include <hip/hip_runtime.h>
#include <hip/hip_bf16.h>

typedef __attribute__((ext_vector_type(16))) __bf16 v16bf;
typedef __attribute__((ext_vector_type(8)))  __bf16 bf16x8;
typedef __attribute__((ext_vector_type(2)))  __bf16 bf16x2;
typedef __attribute__((ext_vector_type(8)))  float  v8f;
typedef __attribute__((ext_vector_type(4)))  float  f32x4;
typedef __attribute__((ext_vector_type(2)))  float  f32x2;

constexpr int   N_NODES = 8192;
constexpr int   NFEAT   = 512;
constexpr int   NHID    = 256;
constexpr int   NZ      = 64;
constexpr int   NEDGE   = 131072;
constexpr float LRELU_A = 0.2f;

__device__ __forceinline__ v8f wmma_bf16(v16bf a, v16bf b, v8f c) {
  return __builtin_amdgcn_wmma_f32_16x16x32_bf16(false, a, false, b, (short)0, c,
                                                 false, false);
}

// Load a 16-element bf16 fragment from one padded LDS row (2x ds_load_b128).
// Per ISA 7.12.2: lane<16 -> K = 0..7 & 16..23 ; lane>=16 -> K = 8..15 & 24..31,
// packed as elem[0..7] = K(half..half+7), elem[8..15] = K(16+half..16+half+7).
__device__ __forceinline__ v16bf load_frag16(const __bf16* row, int khalf) {
  bf16x8 lo = *(const bf16x8*)(row + khalf);
  bf16x8 hi = *(const bf16x8*)(row + 16 + khalf);
  v16bf r;
#pragma unroll
  for (int t = 0; t < 8; ++t) { r[t] = lo[t]; r[8 + t] = hi[t]; }
  return r;
}

__device__ __forceinline__ float act_apply(float x, int act) {
  if (act == 1) return fmaxf(x, 0.0f);                 // relu
  if (act == 2) return 1.0f / (1.0f + __expf(-x));     // sigmoid
  return x;
}

// ---------------------------------------------------------------------------
// Generic GEMM: C[M,N] = act(A @ B + bias).  A is split at column K1: k < K1
// from A1 (ld=K1), else from A2 (ld=K-K1).  For unsplit pass A2=A1, K1=K.
// K1 % 32 == 0, so each 32-wide K tile comes entirely from one source ->
// pointer selection hoisted out of the staging loop (branch-free inner loop).
// Block = 256 threads = 8 waves -> 64x64 tile; wave (wm,wn) = 16 rows x 32 cols.
// ---------------------------------------------------------------------------
__global__ __launch_bounds__(256) void k_gemm(
    const float* __restrict__ A1, const float* __restrict__ A2, int K1,
    const float* __restrict__ B, const float* __restrict__ bias,
    float* __restrict__ C, int M, int N, int K, int act) {
  __shared__ __align__(16) __bf16 As[64][40];    // row-major [m][k]
  __shared__ __align__(16) __bf16 BsT[64][40];   // K-transposed [n][k]

  const int tid  = threadIdx.x;
  const int lane = tid & 31;
  const int wave = tid >> 5;
  const int wm = wave & 3, wn = wave >> 2;
  const int m0 = blockIdx.y * 64, n0 = blockIdx.x * 64;
  const int khalf = (lane < 16) ? 0 : 8;
  const int nl = lane & 15;

  // staging coordinates
  const int ar = tid >> 2, akk = (tid & 3) * 8;       // A: row, 8 consecutive k
  const int bc = tid & 63, brb = (tid >> 6) * 8;      // B: col, 8 consecutive k

  v8f acc0 = {}, acc1 = {};

  for (int k0 = 0; k0 < K; k0 += 32) {
    // ---- tile-uniform A-source selection (no per-element branches) ----
    const bool lo = (k0 < K1);
    const float* Ap = lo ? A1 : A2;
    const int lda   = lo ? K1 : (K - K1);
    const int kbase = lo ? k0 : (k0 - K1);

    { // stage A: two float4 loads -> packed bf16x8 -> one ds_store_b128
      const float* src = Ap + (size_t)(m0 + ar) * lda + kbase + akk;
      f32x4 v0 = *(const f32x4*)src;
      f32x4 v1 = *(const f32x4*)(src + 4);
      bf16x8 pk;
#pragma unroll
      for (int t = 0; t < 4; ++t) {
        pk[t]     = (__bf16)v0[t];
        pk[4 + t] = (__bf16)v1[t];
      }
      *(bf16x8*)&As[ar][akk] = pk;
    }
    { // stage B transposed: 8 coalesced b32 loads -> one ds_store_b128
      bf16x8 pk;
#pragma unroll
      for (int u = 0; u < 8; ++u)
        pk[u] = (__bf16)B[(size_t)(k0 + brb + u) * N + n0 + bc];
      *(bf16x8*)&BsT[bc][brb] = pk;
    }
    __syncthreads();
    v16bf a  = load_frag16(&As[wm * 16 + nl][0], khalf);
    v16bf b0 = load_frag16(&BsT[wn * 32 + nl][0], khalf);
    v16bf b1 = load_frag16(&BsT[wn * 32 + 16 + nl][0], khalf);
    acc0 = wmma_bf16(a, b0, acc0);
    acc1 = wmma_bf16(a, b1, acc1);
    __syncthreads();
  }

  const int mbase = m0 + wm * 16 + ((lane < 16) ? 0 : 8);
  const int c0 = n0 + wn * 32 + nl, c1 = c0 + 16;
  const float bv0 = bias ? bias[c0] : 0.0f;
  const float bv1 = bias ? bias[c1] : 0.0f;
#pragma unroll
  for (int v = 0; v < 8; ++v) {
    C[(size_t)(mbase + v) * N + c0] = act_apply(acc0[v] + bv0, act);
    C[(size_t)(mbase + v) * N + c1] = act_apply(acc1[v] + bv1, act);
  }
}

// ---------------------------------------------------------------------------
// GAT attention logit projections: f1[i] = Wh[i,:]·a[:F], f2[i] = Wh[i,:]·a[F:]
// ---------------------------------------------------------------------------
__global__ __launch_bounds__(256) void k_f12(const float* __restrict__ Wh,
                                             const float* __restrict__ a,
                                             float* __restrict__ f1,
                                             float* __restrict__ f2, int F) {
  const int lane = threadIdx.x & 31;
  const int i = blockIdx.x * 8 + (threadIdx.x >> 5);
  float s1 = 0.f, s2 = 0.f;
  for (int k = lane; k < F; k += 32) {
    float v = Wh[(size_t)i * F + k];
    s1 += v * a[k];
    s2 += v * a[F + k];
  }
#pragma unroll
  for (int off = 16; off > 0; off >>= 1) {
    s1 += __shfl_down(s1, off, 32);
    s2 += __shfl_down(s2, off, 32);
  }
  if (lane == 0) { f1[i] = s1; f2[i] = s2; }
}

// ---------------------------------------------------------------------------
// Streaming online-softmax row statistics for masked GAT attention.
// One wave per row; never materializes e.
// ---------------------------------------------------------------------------
__global__ __launch_bounds__(256) void k_rowstats(
    const float* __restrict__ adj, const float* __restrict__ f1,
    const float* __restrict__ f2, float* __restrict__ rmax,
    float* __restrict__ rsum, int N) {
  const int lane = threadIdx.x & 31;
  const int i = blockIdx.x * 8 + (threadIdx.x >> 5);
  const float fi = f1[i];
  float m = -3.0e38f, s = 0.0f;
  for (int j = lane; j < N; j += 32) {
    float av = adj[(size_t)i * N + j];
    float e = fi + f2[j];
    e = (e > 0.f) ? e : LRELU_A * e;
    float val = (av > 0.f) ? e : -1e9f;
    float nm = fmaxf(m, val);
    s = s * __expf(m - nm) + __expf(val - nm);
    m = nm;
  }
#pragma unroll
  for (int off = 16; off > 0; off >>= 1) {
    float m2 = __shfl_down(m, off, 32), s2 = __shfl_down(s, off, 32);
    float nm = fmaxf(m, m2);
    s = s * __expf(m - nm) + s2 * __expf(m2 - nm);
    m = nm;
  }
  if (lane == 0) { rmax[i] = m; rsum[i] = s; }
}

// ---------------------------------------------------------------------------
// Fused aggregation: Out[16 x F] = epilogue( P @ Y ), streaming P row-block.
// mode 0: P = softmax weights exp(e-max) recomputed on the fly, epi = ELU(x/sum)
// mode 1: P = adj, epi = relu(x + bias) ; mode 2: P = adj, epi = x + bias
// F == 256 (= blockDim). 16 rows per WG; each wave owns a 32-col slice of F.
// Y panel (8 MB) is L2-resident (192 MB L2); adj streams exactly once.
// ---------------------------------------------------------------------------
__global__ __launch_bounds__(256) void k_aggregate(
    const float* __restrict__ adj, const float* __restrict__ Y,
    const float* __restrict__ f1, const float* __restrict__ f2,
    const float* __restrict__ rowmax, const float* __restrict__ rowsum,
    const float* __restrict__ bias, float* __restrict__ Out, int N, int F,
    int mode) {
  __shared__ __align__(16) __bf16 Ps[16][40];     // [i][j]
  __shared__ __align__(16) __bf16 YsT[256][40];   // [f][j]

  const int tid  = threadIdx.x;
  const int lane = tid & 31;
  const int wave = tid >> 5;
  const int i0 = blockIdx.x * 16;
  const int fc = wave * 32;
  const int khalf = (lane < 16) ? 0 : 8;
  const int nl = lane & 15;

  // P staging: thread -> row pi, adjacent column pair pj (packed b32 store)
  const int pi = tid >> 4;
  const int pj = (tid & 15) * 2;
  float f1v = 0.f, mv = 0.f;
  if (mode == 0) { f1v = f1[i0 + pi]; mv = rowmax[i0 + pi]; }

  v8f acc0 = {}, acc1 = {};

  for (int j0 = 0; j0 < N; j0 += 32) {
    if (j0 + 32 < N)
      __builtin_prefetch(&adj[(size_t)(i0 + pi) * N + j0 + 32 + pj], 0, 0);
    {
      f32x2 av = *(const f32x2*)&adj[(size_t)(i0 + pi) * N + j0 + pj];
      float p0, p1;
      if (mode == 0) {
        float e0 = f1v + f2[j0 + pj];
        float e1 = f1v + f2[j0 + pj + 1];
        e0 = (e0 > 0.f) ? e0 : LRELU_A * e0;
        e1 = (e1 > 0.f) ? e1 : LRELU_A * e1;
        p0 = (av[0] > 0.f) ? __expf(e0 - mv) : 0.0f;
        p1 = (av[1] > 0.f) ? __expf(e1 - mv) : 0.0f;
      } else { p0 = av[0]; p1 = av[1]; }
      bf16x2 pp; pp[0] = (__bf16)p0; pp[1] = (__bf16)p1;
      *(bf16x2*)&Ps[pi][pj] = pp;
    }
    // stage Y panel transposed: coalesced b32 loads -> 4x ds_store_b128
#pragma unroll
    for (int rq = 0; rq < 4; ++rq) {
      bf16x8 pk;
#pragma unroll
      for (int u = 0; u < 8; ++u)
        pk[u] = (__bf16)Y[(size_t)(j0 + rq * 8 + u) * F + tid];
      *(bf16x8*)&YsT[tid][rq * 8] = pk;
    }
    __syncthreads();
    v16bf a  = load_frag16(&Ps[nl][0], khalf);
    v16bf b0 = load_frag16(&YsT[fc + nl][0], khalf);
    v16bf b1 = load_frag16(&YsT[fc + 16 + nl][0], khalf);
    acc0 = wmma_bf16(a, b0, acc0);
    acc1 = wmma_bf16(a, b1, acc1);
    __syncthreads();
  }

  const int mrow = (lane < 16) ? 0 : 8;
  const int c0 = fc + nl, c1 = c0 + 16;
#pragma unroll
  for (int v = 0; v < 8; ++v) {
    int gi = i0 + mrow + v;
    float x0 = acc0[v], x1 = acc1[v];
    if (mode == 0) {
      float inv = 1.0f / rowsum[gi];
      x0 *= inv; x1 *= inv;
      x0 = (x0 > 0.f) ? x0 : __expf(x0) - 1.0f;   // ELU (concat=True)
      x1 = (x1 > 0.f) ? x1 : __expf(x1) - 1.0f;
    } else {
      x0 += bias[c0]; x1 += bias[c1];
      if (mode == 1) { x0 = fmaxf(x0, 0.f); x1 = fmaxf(x1, 0.f); }
    }
    Out[(size_t)gi * F + c0] = x0;
    Out[(size_t)gi * F + c1] = x1;
  }
}

// ---------------------------------------------------------------------------
// z = mu + eps*exp(logvar/2); per-row unbiased variance -> var[i]
// ---------------------------------------------------------------------------
__global__ __launch_bounds__(256) void k_reparam(
    const float* __restrict__ mu, const float* __restrict__ lv,
    const float* __restrict__ eps, float* __restrict__ z,
    float* __restrict__ var) {
  const int lane = threadIdx.x & 31;
  const int i = blockIdx.x * 8 + (threadIdx.x >> 5);
  float zs = 0.f, zq = 0.f;
#pragma unroll
  for (int u = 0; u < 2; ++u) {
    size_t idx = (size_t)i * NZ + lane + 32 * u;
    float v = mu[idx] + eps[idx] * __expf(0.5f * lv[idx]);
    z[idx] = v;
    zs += v; zq += v * v;
  }
#pragma unroll
  for (int off = 16; off > 0; off >>= 1) {
    zs += __shfl_down(zs, off, 32);
    zq += __shfl_down(zq, off, 32);
  }
  if (lane == 0) var[i] = (zq - zs * zs / (float)NZ) / (float)(NZ - 1);
}

__global__ __launch_bounds__(256) void k_finalize(
    const float* __restrict__ z, const float* __restrict__ var,
    float* __restrict__ zmean, float* __restrict__ zsum) {
  __shared__ float sred[256];
  const int t = threadIdx.x;
  if (t < NZ) {
    float s = 0.f;
    for (int i = 0; i < N_NODES; ++i) s += z[(size_t)i * NZ + t];
    zmean[t] = s / (float)N_NODES;
  }
  float p = 0.f;
  for (int i = t; i < N_NODES; i += 256) p += var[i];
  sred[t] = p;
  __syncthreads();
  for (int o = 128; o > 0; o >>= 1) {
    if (t < o) sred[t] += sred[t + o];
    __syncthreads();
  }
  if (t == 0) zsum[0] = sred[0];
}

// ---------------------------------------------------------------------------
// Edge MLP, fully fused: gather [X[src]||S[src] ; X[dst]||S[dst]] (K=640),
// WMMA @ l1_w, relu(+b1), then fold the l2 GEMV into the epilogue.
// 16 edges per WG; 8 waves cover the full 256-wide hidden dim.
// ---------------------------------------------------------------------------
__global__ __launch_bounds__(256) void k_edge(
    const int* __restrict__ ids, const float* __restrict__ X,
    const float* __restrict__ S, const float* __restrict__ W1,
    const float* __restrict__ b1, const float* __restrict__ w2,
    const float* __restrict__ b2, float* __restrict__ outp) {
  __shared__ __align__(16) __bf16 As[16][40];
  __shared__ __align__(16) __bf16 BsT[256][40];
  __shared__ int sid[32];
  __shared__ float red[8][32][8];

  const int tid  = threadIdx.x;
  const int lane = tid & 31;
  const int wave = tid >> 5;
  const int e0 = blockIdx.x * 16;
  const int fc = wave * 32;
  const int khalf = (lane < 16) ? 0 : 8;
  const int nl = lane & 15;

  // gather staging: thread -> edge row gm, adjacent k pair gkp
  const int gm = tid >> 4;
  const int gkp = (tid & 15) * 2;

  if (tid < 32) sid[tid] = ids[e0 * 2 + tid];
  __syncthreads();

  v8f acc0 = {}, acc1 = {};

  for (int k0 = 0; k0 < 640; k0 += 32) {
    { // A gather: segment boundaries (320, 256) are even -> pairs never straddle
      float vv[2];
#pragma unroll
      for (int u = 0; u < 2; ++u) {
        int gk = k0 + gkp + u;
        int which = (gk >= 320) ? 1 : 0;
        int node = sid[gm * 2 + which];
        int c = gk - which * 320;
        vv[u] = (c < 256) ? X[(size_t)node * 256 + c]
                          : S[(size_t)node * 64 + (c - 256)];
      }
      bf16x2 pp; pp[0] = (__bf16)vv[0]; pp[1] = (__bf16)vv[1];
      *(bf16x2*)&As[gm][gkp] = pp;
    }
    // W1 panel transposed: coalesced b32 loads -> 4x ds_store_b128
#pragma unroll
    for (int rq = 0; rq < 4; ++rq) {
      bf16x8 pk;
#pragma unroll
      for (int u = 0; u < 8; ++u)
        pk[u] = (__bf16)W1[(size_t)(k0 + rq * 8 + u) * 256 + tid];
      *(bf16x8*)&BsT[tid][rq * 8] = pk;
    }
    __syncthreads();
    v16bf a   = load_frag16(&As[nl][0], khalf);
    v16bf b0  = load_frag16(&BsT[fc + nl][0], khalf);
    v16bf b1f = load_frag16(&BsT[fc + 16 + nl][0], khalf);
    acc0 = wmma_bf16(a, b0, acc0);
    acc1 = wmma_bf16(a, b1f, acc1);
    __syncthreads();
  }

  const int c0 = fc + nl, c1 = c0 + 16;
  const float w20 = w2[c0], w21 = w2[c1];
  const float bb0 = b1[c0], bb1 = b1[c1];
#pragma unroll
  for (int v = 0; v < 8; ++v)
    red[wave][lane][v] =
        fmaxf(acc0[v] + bb0, 0.f) * w20 + fmaxf(acc1[v] + bb1, 0.f) * w21;
  __syncthreads();
  if (tid < 16) {
    float s = b2[0];
    int laneLo = (tid < 8) ? 0 : 16;
    int v = tid & 7;
    for (int w = 0; w < 8; ++w)
      for (int l = 0; l < 16; ++l) s += red[w][laneLo + l][v];
    outp[e0 + tid] = s;
  }
}

// ---------------------------------------------------------------------------
extern "C" void kernel_launch(void* const* d_in, const int* in_sizes, int n_in,
                              void* d_out, int out_size, void* d_ws,
                              size_t ws_size, hipStream_t stream) {
  const float* feat   = (const float*)d_in[0];
  const float* adj    = (const float*)d_in[1];
  const int*   ids    = (const int*)  d_in[2];
  const float* eps    = (const float*)d_in[3];
  const float* az_W   = (const float*)d_in[4];
  const float* az_a   = (const float*)d_in[5];
  const float* as_W   = (const float*)d_in[6];
  const float* as_a   = (const float*)d_in[7];
  const float* enc_w1 = (const float*)d_in[8];
  const float* enc_b1 = (const float*)d_in[9];
  const float* enc_w2 = (const float*)d_in[10];
  const float* enc_b2 = (const float*)d_in[11];
  const float* lins_w = (const float*)d_in[12];
  const float* lins_b = (const float*)d_in[13];
  const float* dec_w  = (const float*)d_in[14];
  const float* dec_b  = (const float*)d_in[15];
  const float* gc1_w  = (const float*)d_in[16];
  const float* gc1_b  = (const float*)d_in[17];
  const float* gc2_w  = (const float*)d_in[18];
  const float* gc2_b  = (const float*)d_in[19];
  const float* l1_w   = (const float*)d_in[20];
  const float* l1_b   = (const float*)d_in[21];
  const float* l2_w   = (const float*)d_in[22];
  const float* l2_b   = (const float*)d_in[23];
  (void)in_sizes; (void)n_in; (void)out_size; (void)ws_size;

  // Output layout (flat, reference return order)
  float* out    = (float*)d_out;
  float* zmean  = out;                                    // [64]
  float* outEdge= out + 64;                               // [E]
  float* recon  = outEdge + NEDGE;                        // [N,512]
  float* mu     = recon + (size_t)N_NODES * NFEAT;        // [N,64]
  float* logv   = mu + (size_t)N_NODES * NZ;              // [N,64]
  float* zsum   = logv + (size_t)N_NODES * NZ;            // [1]

  // Workspace
  float* w = (float*)d_ws;
  size_t o = 0;
  float* WhZ = w + o; o += (size_t)N_NODES * NHID;
  float* WhS = w + o; o += (size_t)N_NODES * NHID;
  float* HZ  = w + o; o += (size_t)N_NODES * NHID;
  float* HS  = w + o; o += (size_t)N_NODES * NHID;
  float* Y   = w + o; o += (size_t)N_NODES * NHID;
  float* X1  = w + o; o += (size_t)N_NODES * NHID;
  float* X2  = w + o; o += (size_t)N_NODES * NHID;
  float* Z   = w + o; o += (size_t)N_NODES * NZ;
  float* Sb  = w + o; o += (size_t)N_NODES * NZ;
  float* f1z = w + o; o += N_NODES;
  float* f2z = w + o; o += N_NODES;
  float* f1s = w + o; o += N_NODES;
  float* f2s = w + o; o += N_NODES;
  float* mxz = w + o; o += N_NODES;
  float* smz = w + o; o += N_NODES;
  float* mxs = w + o; o += N_NODES;
  float* sms = w + o; o += N_NODES;
  float* var = w + o; o += N_NODES;

  const dim3 blk(256);
  const int MR = N_NODES / 64;  // 128 row tiles for node GEMMs

  // --- GAT projections Wh = feat @ W ---
  k_gemm<<<dim3(NHID / 64, MR), blk, 0, stream>>>(feat, feat, NFEAT, az_W,
      nullptr, WhZ, N_NODES, NHID, NFEAT, 0);
  k_gemm<<<dim3(NHID / 64, MR), blk, 0, stream>>>(feat, feat, NFEAT, as_W,
      nullptr, WhS, N_NODES, NHID, NFEAT, 0);

  // --- attention logits + streaming softmax stats ---
  k_f12<<<N_NODES / 8, blk, 0, stream>>>(WhZ, az_a, f1z, f2z, NHID);
  k_f12<<<N_NODES / 8, blk, 0, stream>>>(WhS, as_a, f1s, f2s, NHID);
  k_rowstats<<<N_NODES / 8, blk, 0, stream>>>(adj, f1z, f2z, mxz, smz, N_NODES);
  k_rowstats<<<N_NODES / 8, blk, 0, stream>>>(adj, f1s, f2s, mxs, sms, N_NODES);

  // --- fused masked-softmax aggregation + ELU ---
  k_aggregate<<<N_NODES / 16, blk, 0, stream>>>(adj, WhZ, f1z, f2z, mxz, smz,
      nullptr, HZ, N_NODES, NHID, 0);
  k_aggregate<<<N_NODES / 16, blk, 0, stream>>>(adj, WhS, f1s, f2s, mxs, sms,
      nullptr, HS, N_NODES, NHID, 0);

  // --- encoder (mu, logvar directly into d_out) ---
  k_gemm<<<dim3(NZ / 64, MR), blk, 0, stream>>>(HZ, HZ, NHID, enc_w1,
      enc_b1, mu, N_NODES, NZ, NHID, 1);
  k_gemm<<<dim3(NZ / 64, MR), blk, 0, stream>>>(HZ, HZ, NHID, enc_w2,
      enc_b2, logv, N_NODES, NZ, NHID, 1);

  // --- reparametrize + stats ---
  k_reparam<<<N_NODES / 8, blk, 0, stream>>>(mu, logv, eps, Z, var);
  k_finalize<<<1, blk, 0, stream>>>(Z, var, zmean, zsum);

  // --- s = relu([z || h_s] @ lins_w + b)   (split-A GEMM, K = 64 + 256) ---
  k_gemm<<<dim3(NZ / 64, MR), blk, 0, stream>>>(Z, HS, NZ, lins_w, lins_b, Sb,
      N_NODES, NZ, NZ + NHID, 1);

  // --- decoder: recon_x = sigmoid(s @ dec_w + b) into d_out ---
  k_gemm<<<dim3(NFEAT / 64, MR), blk, 0, stream>>>(Sb, Sb, NZ, dec_w,
      dec_b, recon, N_NODES, NFEAT, NZ, 2);

  // --- GCN layer 1: x1 = relu(adj @ (recon @ gc1_w) + b) ---
  k_gemm<<<dim3(NHID / 64, MR), blk, 0, stream>>>(recon, recon, NFEAT, gc1_w,
      nullptr, Y, N_NODES, NHID, NFEAT, 0);
  k_aggregate<<<N_NODES / 16, blk, 0, stream>>>(adj, Y, nullptr, nullptr,
      nullptr, nullptr, gc1_b, X1, N_NODES, NHID, 1);

  // --- GCN layer 2: x2 = adj @ (x1 @ gc2_w) + b ---
  k_gemm<<<dim3(NHID / 64, MR), blk, 0, stream>>>(X1, X1, NHID, gc2_w,
      nullptr, Y, N_NODES, NHID, NHID, 0);
  k_aggregate<<<N_NODES / 16, blk, 0, stream>>>(adj, Y, nullptr, nullptr,
      nullptr, nullptr, gc2_b, X2, N_NODES, NHID, 2);

  // --- fused edge gather -> l1 (WMMA) -> relu -> l2 GEMV ---
  k_edge<<<NEDGE / 16, blk, 0, stream>>>(ids, X2, Sb, l1_w, l1_b, l2_w, l2_b,
      outEdge);
}
